// MeanStdPooling_70935679861209
// MI455X (gfx1250) — matrix operand order repaired and checked by reading, not compiled, not run on hardware
//
#include <hip/hip_runtime.h>
#include <hip/hip_bf16.h>

// ---------------------------------------------------------------------------
// MI455X (gfx1250) fused MLP chain:
//   h = elu(x@W1+b1); h = elu(h@W2+b2); m = max(child,parent_gather);
//   r = elu(m@W3+b3); r = r@W4+b4  -> (r[:,0], r[:,1])
// All GEMMs on v_wmma_f32_16x16x32_f16; weights staged in LDS via TDM;
// next-tile operands prefetched with global_prefetch_b8.
// ---------------------------------------------------------------------------

typedef __attribute__((ext_vector_type(16))) _Float16 v16h;
typedef __attribute__((ext_vector_type(8)))  _Float16 v8h;
typedef __attribute__((ext_vector_type(8)))  float    v8f;
typedef __attribute__((ext_vector_type(4)))  float    v4f;
typedef __attribute__((ext_vector_type(4)))  unsigned u32x4;
typedef __attribute__((ext_vector_type(8)))  int      i32x8;
typedef __attribute__((ext_vector_type(4)))  int      i32x4;

#define BN_ROWS   262016      // B*N = 128*2047
#define M2_ROWS   261888      // B*(N-1) = 128*2046
#define TILES1    16376       // BN_ROWS/16
#define TILES2    16368       // M2_ROWS/16

union V16 { v16h v; v8h h2[2]; };

#if __has_builtin(__builtin_amdgcn_tensor_load_to_lds)
#define HAVE_TDM 1
#else
#define HAVE_TDM 0
#endif

static __device__ __forceinline__ float eluf(float v) {
    return v > 0.0f ? v : (__expf(v) - 1.0f);
}

static __device__ __forceinline__ v8f wmma16(const V16& a, const V16& b, v8f c) {
    return __builtin_amdgcn_wmma_f32_16x16x32_f16(
        /*neg_a=*/false, a.v, /*neg_b=*/false, b.v,
        /*c_mod=*/(short)0, c, /*reuse_a=*/false, /*reuse_b=*/false);
}

// B fragment: 16 contiguous halves per lane at ((nt*KT+kt)*32+lane)*32 bytes.
static __device__ __forceinline__ V16 ldsB(const char* base, int nt, int kt, int KT, int lane) {
    const v8h* p = (const v8h*)(base + (((nt * KT + kt) * 32 + lane) << 5));
    V16 r; r.h2[0] = p[0]; r.h2[1] = p[1]; return r;
}

// 1-D TDM copy global -> LDS (D# per CDNA5 ISA ch.8), data_size=8B units.
static __device__ __forceinline__ void tdm_load_1d(unsigned ldsAddr,
                                                   unsigned long long gaddr,
                                                   unsigned nbytes) {
#if HAVE_TDM
    unsigned u = nbytes >> 3;  // 8-byte elements
    u32x4 g0 = { 1u,                                   // count=1, user mode
                 ldsAddr,                              // lds_addr[31:0]
                 (unsigned)(gaddr & 0xFFFFFFFFull),    // global_addr lo
                 (unsigned)((gaddr >> 32) & 0x1FFFFFFull) | (2u << 30) }; // type=2
    i32x8 g1 = { (int)(3u << 16),                      // data_size=3 (8B)
                 (int)((u & 0xFFFFu) << 16),           // tensor_dim0[15:0]
                 (int)(((u >> 16) & 0xFFFFu) | (1u << 16)), // dim0 hi | tensor_dim1=1
                 (int)((u & 0xFFFFu) << 16),           // tile_dim0 = u
                 1,                                    // tile_dim1=1, tile_dim2=0
                 (int)u,                               // tensor_dim0_stride lo
                 0, 0 };
    i32x4 z4 = {0, 0, 0, 0};
#if defined(__clang_major__) && (__clang_major__ >= 23)
    i32x8 z8 = {0, 0, 0, 0, 0, 0, 0, 0};
    __builtin_amdgcn_tensor_load_to_lds(g0, g1, z4, z4, z8, 0);
#else
    __builtin_amdgcn_tensor_load_to_lds(g0, g1, z4, z4, 0);
#endif
#endif
}

// ---------------------------------------------------------------------------
// Pack a (K x Nfull) f32 weight into f16 B-fragment order (Npad columns).
// packed[((nt*(K/32)+kt)*32 + lane)*16 + h] = W[(kt*32+8*(lane>>4)+(h&7)+16*(h>>3))*Nfull + nt*16+(lane&15)]
// ---------------------------------------------------------------------------
__global__ void pack_weights(const float* __restrict__ W, _Float16* __restrict__ dst,
                             int K, int Nfull, int Npad) {
    int total = (Npad >> 4) * (K >> 5) * 512;
    for (int p = blockIdx.x * blockDim.x + threadIdx.x; p < total;
         p += gridDim.x * blockDim.x) {
        int h  = p & 15;
        int l  = (p >> 4) & 31;
        int kt = (p >> 9) % (K >> 5);
        int nt = (p >> 9) / (K >> 5);
        int k  = kt * 32 + ((l >> 4) << 3) + (h & 7) + ((h >> 3) << 4);
        int n  = nt * 16 + (l & 15);
        float v = (n < Nfull) ? W[k * Nfull + n] : 0.0f;
        dst[p] = (_Float16)v;
    }
}

// ---------------------------------------------------------------------------
// Kernel 1: h16 = elu(elu(x@W1+b1)@W2+b2), stored f16 row-major [BN,128].
// LDS: [0,64K) W1 frags, [64K,96K) W2 frags, [96K,112K) 4 x 4KB wave tiles.
// ---------------------------------------------------------------------------
__global__ __launch_bounds__(128)
void encode_mlp(const float* __restrict__ x, const _Float16* __restrict__ wpack,
                const float* __restrict__ b1, const float* __restrict__ b2,
                _Float16* __restrict__ h16, int numTiles) {
    extern __shared__ char smem[];
    char* w1l = smem;
    char* w2l = smem + 65536;
    const int wave = threadIdx.x >> 5;
    const int lane = threadIdx.x & 31;
    char* htile = smem + 98304 + wave * 4096;

#if HAVE_TDM
    if (threadIdx.x == 0) {
        tdm_load_1d((unsigned)(size_t)(void*)smem,
                    (unsigned long long)(size_t)(const void*)wpack, 98304u);
        __builtin_amdgcn_s_wait_tensorcnt(0);
    }
#else
    for (unsigned i = threadIdx.x; i < 98304u / 16u; i += blockDim.x)
        ((v8h*)smem)[i] = ((const v8h*)wpack)[i];
#endif
    __syncthreads();

    const int m  = lane & 15;
    const int hi = lane >> 4;
    const int kb = hi << 3;

    float bi1[8], bi2[8];
#pragma unroll
    for (int nt = 0; nt < 8; ++nt) {
        bi1[nt] = b1[nt * 16 + m];
        bi2[nt] = b2[nt * 16 + m];
    }

    const v8f vzero = {0, 0, 0, 0, 0, 0, 0, 0};
    const int wstride = gridDim.x * (blockDim.x >> 5);

    for (int tile = blockIdx.x * (blockDim.x >> 5) + wave; tile < numTiles;
         tile += wstride) {
        const int rowBase = tile << 4;

        // Prefetch next tile's x row for this lane (global_prefetch_b8).
        if (tile + wstride < numTiles) {
            const float* xn = x + (size_t)(rowBase + wstride * 16 + m) * 256;
            __builtin_prefetch(xn, 0, 1);
            __builtin_prefetch(xn + 128, 0, 1);
        }

        v8f acc[8];
#pragma unroll
        for (int nt = 0; nt < 8; ++nt) acc[nt] = vzero;

        // ---- layer 1: A from x (f32 -> f16), B from LDS, K = 256 ----
        const float* xrow = x + (size_t)(rowBase + m) * 256;
#pragma unroll
        for (int kt = 0; kt < 8; ++kt) {
            const int k0 = kt * 32 + kb;
            v4f f0 = *(const v4f*)(xrow + k0);
            v4f f1 = *(const v4f*)(xrow + k0 + 4);
            v4f f2 = *(const v4f*)(xrow + k0 + 16);
            v4f f3 = *(const v4f*)(xrow + k0 + 20);
            V16 a;
#pragma unroll
            for (int j = 0; j < 4; ++j) {
                a.v[j]      = (_Float16)f0[j];
                a.v[4 + j]  = (_Float16)f1[j];
                a.v[8 + j]  = (_Float16)f2[j];
                a.v[12 + j] = (_Float16)f3[j];
            }
#pragma unroll
            for (int nt = 0; nt < 8; ++nt) {
                V16 b = ldsB(w1l, nt, kt, 8, lane);
                acc[nt] = wmma16(a, b, acc[nt]);
            }
        }

        // ---- epilogue 1: bias+ELU -> wave-private LDS tile (f16) ----
        _Float16* ht = (_Float16*)htile;
#pragma unroll
        for (int nt = 0; nt < 8; ++nt) {
            const int n = nt * 16 + m;
#pragma unroll
            for (int r = 0; r < 8; ++r)
                ht[(r + 8 * hi) * 128 + n] = (_Float16)eluf(acc[nt][r] + bi1[nt]);
            acc[nt] = vzero;
        }

        // ---- layer 2: A from LDS tile, K = 128 (same-wave LDS is in-order) ----
#pragma unroll
        for (int kt = 0; kt < 4; ++kt) {
            const int k0 = kt * 32 + kb;
            V16 a;
            a.h2[0] = *(const v8h*)(htile + ((m * 128 + k0) << 1));
            a.h2[1] = *(const v8h*)(htile + ((m * 128 + k0 + 16) << 1));
#pragma unroll
            for (int nt = 0; nt < 8; ++nt) {
                V16 b = ldsB(w2l, nt, kt, 4, lane);
                acc[nt] = wmma16(a, b, acc[nt]);
            }
        }

        // ---- epilogue 2: bias+ELU -> LDS, then coalesced b128 store ----
#pragma unroll
        for (int nt = 0; nt < 8; ++nt) {
            const int n = nt * 16 + m;
#pragma unroll
            for (int r = 0; r < 8; ++r)
                ht[(r + 8 * hi) * 128 + n] = (_Float16)eluf(acc[nt][r] + bi2[nt]);
        }
        _Float16* hout = h16 + (size_t)rowBase * 128;
#pragma unroll
        for (int r = 0; r < 8; ++r) {
            const int chunk = r * 32 + lane;           // lanes contiguous -> coalesced
            *(v8h*)(hout + chunk * 8) = *(const v8h*)(htile + (chunk << 4));
        }
    }
}

// ---------------------------------------------------------------------------
// Kernel 2: gather child/parent from h16, max, elu(@W3+b3) @ W4 + b4.
// LDS: [0,32K) W3 frags, [32K,36K) W4 frags (N padded to 16), 4 x 4KB tiles.
// ---------------------------------------------------------------------------
__global__ __launch_bounds__(128)
void decode_mlp(const _Float16* __restrict__ h16, const int* __restrict__ pidx,
                const _Float16* __restrict__ w34pack,
                const float* __restrict__ b3, const float* __restrict__ b4,
                float* __restrict__ out, int numTiles) {
    extern __shared__ char smem[];
    char* w3l = smem;
    char* w4l = smem + 32768;
    const int wave = threadIdx.x >> 5;
    const int lane = threadIdx.x & 31;
    char* rtile = smem + 36864 + wave * 4096;

#if HAVE_TDM
    if (threadIdx.x == 0) {
        tdm_load_1d((unsigned)(size_t)(void*)smem,
                    (unsigned long long)(size_t)(const void*)w34pack, 36864u);
        __builtin_amdgcn_s_wait_tensorcnt(0);
    }
#else
    for (unsigned i = threadIdx.x; i < 36864u / 16u; i += blockDim.x)
        ((v8h*)smem)[i] = ((const v8h*)w34pack)[i];
#endif
    __syncthreads();

    const int m  = lane & 15;
    const int hi = lane >> 4;
    const int kb = hi << 3;

    float bi3[8];
#pragma unroll
    for (int nt = 0; nt < 8; ++nt) bi3[nt] = b3[nt * 16 + m];
    const float b4c = (m < 2) ? b4[m] : 0.0f;

    // W4 B-fragments are loop-invariant: keep in registers (4 x 8 VGPRs).
    V16 w4f[4];
#pragma unroll
    for (int kt = 0; kt < 4; ++kt) w4f[kt] = ldsB(w4l, 0, kt, 4, lane);

    const v8f vzero = {0, 0, 0, 0, 0, 0, 0, 0};
    const int wstride = gridDim.x * (blockDim.x >> 5);

    for (int tile = blockIdx.x * (blockDim.x >> 5) + wave; tile < numTiles;
         tile += wstride) {
        const int rowBase = tile << 4;
        const int i  = rowBase + m;                    // row in [0, M2)

        // Prefetch next tile's parent_index entries (global_prefetch_b8).
        if (tile + wstride < numTiles)
            __builtin_prefetch(pidx + i + wstride * 16, 0, 1);

        const int bb = i / 2046;
        const int nn = i - bb * 2046;
        const _Float16* crow = h16 + (size_t)(bb * 2047 + nn) * 128;
        const _Float16* prow = h16 + (size_t)pidx[i] * 128;

        v8f acc[8];
#pragma unroll
        for (int nt = 0; nt < 8; ++nt) acc[nt] = vzero;

        // ---- layer 3: A = max(child,parent) fragments, K = 128 ----
#pragma unroll
        for (int kt = 0; kt < 4; ++kt) {
            const int k0 = kt * 32 + kb;
            v8h c0 = *(const v8h*)(crow + k0);
            v8h c1 = *(const v8h*)(crow + k0 + 16);
            v8h p0 = *(const v8h*)(prow + k0);
            v8h p1 = *(const v8h*)(prow + k0 + 16);
            V16 a;
#pragma unroll
            for (int j = 0; j < 8; ++j) {
                a.v[j]     = (c0[j] > p0[j]) ? c0[j] : p0[j];
                a.v[8 + j] = (c1[j] > p1[j]) ? c1[j] : p1[j];
            }
#pragma unroll
            for (int nt = 0; nt < 8; ++nt) {
                V16 b = ldsB(w3l, nt, kt, 4, lane);
                acc[nt] = wmma16(a, b, acc[nt]);
            }
        }

        // ---- epilogue 3: bias+ELU -> LDS tile (f16) ----
        _Float16* rt = (_Float16*)rtile;
#pragma unroll
        for (int nt = 0; nt < 8; ++nt) {
            const int n = nt * 16 + m;
#pragma unroll
            for (int r = 0; r < 8; ++r)
                rt[(r + 8 * hi) * 128 + n] = (_Float16)eluf(acc[nt][r] + bi3[nt]);
        }

        // ---- layer 4: K = 128, single N-tile (cols 2..15 are zero-padded) ----
        v8f d = vzero;
#pragma unroll
        for (int kt = 0; kt < 4; ++kt) {
            const int k0 = kt * 32 + kb;
            V16 a;
            a.h2[0] = *(const v8h*)(rtile + ((m * 128 + k0) << 1));
            a.h2[1] = *(const v8h*)(rtile + ((m * 128 + k0 + 16) << 1));
            d = wmma16(a, w4f[kt], d);
        }

        // lanes with N<2 hold output columns 0/1; rows = rowBase + r + 8*hi
        if (m < 2) {
#pragma unroll
            for (int r = 0; r < 8; ++r)
                out[(size_t)m * M2_ROWS + rowBase + r + 8 * hi] = d[r] + b4c;
        }
    }
}

// ---------------------------------------------------------------------------
extern "C" void kernel_launch(void* const* d_in, const int* in_sizes, int n_in,
                              void* d_out, int out_size, void* d_ws, size_t ws_size,
                              hipStream_t stream) {
    const float* x    = (const float*)d_in[0];
    const int*   pidx = (const int*)d_in[1];
    const float* W1   = (const float*)d_in[2];
    const float* b1   = (const float*)d_in[3];
    const float* W2   = (const float*)d_in[4];
    const float* b2   = (const float*)d_in[5];
    const float* W3   = (const float*)d_in[6];
    const float* b3   = (const float*)d_in[7];
    const float* W4   = (const float*)d_in[8];
    const float* b4   = (const float*)d_in[9];

    char* ws = (char*)d_ws;
    _Float16* h16 = (_Float16*)ws;                       // 262016*128 f16 = 64 MB
    const size_t H16B = (size_t)BN_ROWS * 128 * 2;
    _Float16* wpk = (_Float16*)(ws + H16B);              // packed weights
    // layout: W1p[32768] | W2p[16384] | W3p[16384] | W4p[2048] halves

    pack_weights<<<64, 256, 0, stream>>>(W1, wpk,          256, 128, 128);
    pack_weights<<<64, 256, 0, stream>>>(W2, wpk + 32768,  128, 128, 128);
    pack_weights<<<64, 256, 0, stream>>>(W3, wpk + 49152,  128, 128, 128);
    pack_weights<<<16, 256, 0, stream>>>(W4, wpk + 65536,  128,   2,  16);

    encode_mlp<<<512, 128, 114688, stream>>>(x, wpk, b1, b2, h16, TILES1);
    decode_mlp<<<512, 128,  53248, stream>>>(h16, pidx, wpk + 49152, b3, b4,
                                             (float*)d_out, TILES2);
}
